// AttentionModel_9363028705594
// MI455X (gfx1250) — matrix operand order repaired and checked by reading, not compiled, not run on hardware
//
#include <hip/hip_runtime.h>
#include <hip/hip_fp16.h>

typedef __attribute__((ext_vector_type(16))) _Float16 v16h;
typedef __attribute__((ext_vector_type(8)))  float    v8f;

#define B_   8
#define F_   40
#define T_   1200
#define H_   512
#define G4_  2048   // 4H
#define DH_  1024   // 2H
#define T2_  299
#define C_   5000
#define U_   40

__device__ __forceinline__ float sigf(float x) { return 1.0f / (1.0f + expf(-x)); }

// ----------------------------------------------------------------------------
// small utility kernels
// ----------------------------------------------------------------------------
__global__ void cvt_f16_kernel(const float* __restrict__ s, _Float16* __restrict__ d, int n) {
  int i = blockIdx.x * blockDim.x + threadIdx.x;
  if (i < n) d[i] = (_Float16)s[i];
}
__global__ void zero_f32_kernel(float* p, int n) {
  int i = blockIdx.x * blockDim.x + threadIdx.x;
  if (i < n) p[i] = 0.0f;
}
__global__ void zero_f16_kernel(_Float16* p, int n) {
  int i = blockIdx.x * blockDim.x + threadIdx.x;
  if (i < n) p[i] = (_Float16)0.0f;
}
__global__ void newlen_kernel(const int* __restrict__ len, int* __restrict__ nl) {
  int i = threadIdx.x;
  if (i < B_) nl[i] = ((len[i] - 1) / 2 - 1) / 2;
}

// ----------------------------------------------------------------------------
// conv front-end:  (B,1,40,1200) -> conv1 s2 -> (B,32,20,599) -> conv2 s2 ->
// (B,32,10,299) -> cnn rows (t*B+b, 320)
// ----------------------------------------------------------------------------
__global__ void conv1_kernel(const float* __restrict__ speech,
                             const float* __restrict__ w, const float* __restrict__ bias,
                             const float* __restrict__ gamma, const float* __restrict__ beta,
                             const float* __restrict__ mean, const float* __restrict__ var,
                             float* __restrict__ a1) {
  const int W1 = 599, H1 = 20;
  int idx = blockIdx.x * blockDim.x + threadIdx.x;
  if (idx >= B_ * 32 * H1 * W1) return;
  int wI = idx % W1; int tmp = idx / W1;
  int hI = tmp % H1; tmp /= H1;
  int cI = tmp % 32; int bI = tmp / 32;
  float acc = bias[cI];
  #pragma unroll
  for (int kh = 0; kh < 3; ++kh) {
    int f = 2 * hI - 1 + kh;
    if (f < 0 || f >= F_) continue;
    #pragma unroll
    for (int kw = 0; kw < 3; ++kw) {
      int t = 2 * wI + kw;   // always < 1200
      acc += speech[(size_t)(bI * T_ + t) * F_ + f] * w[(cI * 3 + kh) * 3 + kw];
    }
  }
  float v = (acc - mean[cI]) * (gamma[cI] * rsqrtf(var[cI] + 1e-5f)) + beta[cI];
  a1[idx] = fmaxf(v, 0.0f);
}

__global__ void conv2_kernel(const float* __restrict__ a1,
                             const float* __restrict__ w, const float* __restrict__ bias,
                             const float* __restrict__ gamma, const float* __restrict__ beta,
                             const float* __restrict__ mean, const float* __restrict__ var,
                             float* __restrict__ cnn, _Float16* __restrict__ cnn_h) {
  const int W2 = T2_, H2 = 10, W1 = 599, H1 = 20;
  int idx = blockIdx.x * blockDim.x + threadIdx.x;
  if (idx >= B_ * 32 * H2 * W2) return;
  int wI = idx % W2; int tmp = idx / W2;
  int hI = tmp % H2; tmp /= H2;
  int cI = tmp % 32; int bI = tmp / 32;
  float acc = bias[cI];
  for (int ci = 0; ci < 32; ++ci) {
    #pragma unroll
    for (int kh = 0; kh < 3; ++kh) {
      int f = 2 * hI - 1 + kh;
      if (f < 0 || f >= H1) continue;
      #pragma unroll
      for (int kw = 0; kw < 3; ++kw) {
        int t = 2 * wI + kw;   // < 599
        acc += a1[((size_t)(bI * 32 + ci) * H1 + f) * W1 + t] * w[((cI * 32 + ci) * 3 + kh) * 3 + kw];
      }
    }
  }
  float v = (acc - mean[cI]) * (gamma[cI] * rsqrtf(var[cI] + 1e-5f)) + beta[cI];
  v = fmaxf(v, 0.0f);
  size_t o = (size_t)(wI * B_ + bI) * 320 + cI * 10 + hI;   // (t*B+b, c*10+h)
  cnn[o] = v;
  cnn_h[o] = (_Float16)v;
}

// ----------------------------------------------------------------------------
// Generic WMMA GEMM:  C(MxN) = A_h(MxK) @ W_h(N x K)^T + bias
// 256 threads = 8 waves; each wave owns a 16x64 C strip (4 N-tiles). All four
// B tiles live in DISTINCT registers and are loaded before the 4-wmma burst,
// so VMEM stays in flight across the matrix ops (no full loadcnt drains).
// Out-of-range rows/cols are clamped; stores are guarded.
// ----------------------------------------------------------------------------
__global__ __launch_bounds__(256)
void gemm_wmma_kernel(const _Float16* __restrict__ A, int lda,
                      const _Float16* __restrict__ W, int ldw,
                      float* __restrict__ C, int ldc,
                      _Float16* __restrict__ Ch, int ldch,
                      const float* __restrict__ bias,
                      int M, int N, int K) {
  const int tid = threadIdx.x;
  const int wave = tid >> 5, lane = tid & 31;
  const int n0 = (blockIdx.x * 8 + wave) * 64;
  const int m0 = blockIdx.y * 16;
  if (n0 >= N) return;
  const int nl = lane & 15;
  const int hi = lane >> 4;           // 0 or 1
  const int am = min(m0 + nl, M - 1);             // clamped A row
  const _Float16* arow = A + (size_t)am * lda;

  const _Float16* brow0 = W + (size_t)min(n0 +      nl, N - 1) * ldw;
  const _Float16* brow1 = W + (size_t)min(n0 + 16 + nl, N - 1) * ldw;
  const _Float16* brow2 = W + (size_t)min(n0 + 32 + nl, N - 1) * ldw;
  const _Float16* brow3 = W + (size_t)min(n0 + 48 + nl, N - 1) * ldw;

  v8f acc[4];
  #pragma unroll
  for (int g = 0; g < 4; ++g)
    #pragma unroll
    for (int r = 0; r < 8; ++r) acc[g][r] = 0.0f;

  for (int k0 = 0; k0 < K; k0 += 32) {
    const _Float16* ap = arow + k0 + hi * 16;
    const _Float16* p0 = brow0 + k0 + hi * 16;
    const _Float16* p1 = brow1 + k0 + hi * 16;
    const _Float16* p2 = brow2 + k0 + hi * 16;
    const _Float16* p3 = brow3 + k0 + hi * 16;
    v16h a, b0, b1, b2, b3;
    #pragma unroll
    for (int j = 0; j < 8; ++j) { a[j] = ap[j]; a[j + 8] = ap[8 + j]; }
    #pragma unroll
    for (int j = 0; j < 16; ++j) b0[j] = p0[j];
    #pragma unroll
    for (int j = 0; j < 16; ++j) b1[j] = p1[j];
    #pragma unroll
    for (int j = 0; j < 16; ++j) b2[j] = p2[j];
    #pragma unroll
    for (int j = 0; j < 16; ++j) b3[j] = p3[j];
    __builtin_prefetch(ap + 32, 0, 3);            // speculative; dropped if OOB
    __builtin_prefetch(p0 + 32, 0, 3);
    __builtin_prefetch(p1 + 32, 0, 3);
    __builtin_prefetch(p2 + 32, 0, 3);
    __builtin_prefetch(p3 + 32, 0, 3);
    acc[0] = __builtin_amdgcn_wmma_f32_16x16x32_f16(false, a, false, b0, (short)0, acc[0], false, false);
    acc[1] = __builtin_amdgcn_wmma_f32_16x16x32_f16(false, a, false, b1, (short)0, acc[1], false, false);
    acc[2] = __builtin_amdgcn_wmma_f32_16x16x32_f16(false, a, false, b2, (short)0, acc[2], false, false);
    acc[3] = __builtin_amdgcn_wmma_f32_16x16x32_f16(false, a, false, b3, (short)0, acc[3], false, false);
  }

  const int mb = m0 + hi * 8;
  #pragma unroll
  for (int g = 0; g < 4; ++g) {
    const int col = n0 + g * 16 + nl;
    if (col >= N) continue;
    const float bv = bias ? bias[col] : 0.0f;
    #pragma unroll
    for (int r = 0; r < 8; ++r) {
      int row = mb + r;
      if (row < M) {
        float v = acc[g][r] + bv;
        C[(size_t)row * ldc + col] = v;
        if (Ch) Ch[(size_t)row * ldch + col] = (_Float16)v;
      }
    }
  }
}

// ----------------------------------------------------------------------------
// BiLSTM recurrent scan: one block per direction, 32 waves.
// Wave w owns units [w*16, w*16+16); it computes all 4 gate tiles for those
// units (n = g*512 + u0) so the cell update is wave-local. h (f16, padded to
// 16 rows) lives in LDS; c-state lives in lane registers for all 299 steps.
// All 4 gate B tiles are loaded into distinct registers before the wmma burst.
// ----------------------------------------------------------------------------
__global__ __launch_bounds__(1024)
void lstm_scan_kernel(const float* __restrict__ xw_fw, const float* __restrict__ xw_bw,
                      const _Float16* __restrict__ Whh_fw, const _Float16* __restrict__ Whh_bw,
                      const int* __restrict__ newlen,
                      float* __restrict__ hs, _Float16* __restrict__ hs_h) {
  const int dir = blockIdx.x;
  const _Float16* Whh = dir ? Whh_bw : Whh_fw;
  const float* xw = dir ? xw_bw : xw_fw;
  __shared__ _Float16 h_pad[16 * H_];
  const int tid = threadIdx.x;
  const int wave = tid >> 5, lane = tid & 31;
  const int u0 = wave * 16;
  const int nl = lane & 15;
  const int hi = lane >> 4;

  for (int i = tid; i < 16 * H_; i += 1024) h_pad[i] = (_Float16)0.0f;

  const _Float16* wrow0 = Whh + (size_t)(0 * H_ + u0 + nl) * H_;
  const _Float16* wrow1 = Whh + (size_t)(1 * H_ + u0 + nl) * H_;
  const _Float16* wrow2 = Whh + (size_t)(2 * H_ + u0 + nl) * H_;
  const _Float16* wrow3 = Whh + (size_t)(3 * H_ + u0 + nl) * H_;

  float c_reg[8], h_reg[8];
  int nlen[8];
  #pragma unroll
  for (int r = 0; r < 8; ++r) { c_reg[r] = 0.0f; h_reg[r] = 0.0f; nlen[r] = newlen[r]; }
  __syncthreads();

  for (int s = 0; s < T2_; ++s) {
    const int t = dir ? (T2_ - 1 - s) : s;

    v8f acc[4];
    #pragma unroll
    for (int g = 0; g < 4; ++g)
      #pragma unroll
      for (int r = 0; r < 8; ++r) acc[g][r] = 0.0f;

    for (int k0 = 0; k0 < H_; k0 += 32) {
      const _Float16* ap = &h_pad[(size_t)nl * H_ + k0 + hi * 16];
      const _Float16* p0 = wrow0 + k0 + hi * 16;
      const _Float16* p1 = wrow1 + k0 + hi * 16;
      const _Float16* p2 = wrow2 + k0 + hi * 16;
      const _Float16* p3 = wrow3 + k0 + hi * 16;
      v16h a, b0, b1, b2, b3;
      #pragma unroll
      for (int j = 0; j < 8; ++j) { a[j] = ap[j]; a[j + 8] = ap[8 + j]; }
      #pragma unroll
      for (int j = 0; j < 16; ++j) b0[j] = p0[j];
      #pragma unroll
      for (int j = 0; j < 16; ++j) b1[j] = p1[j];
      #pragma unroll
      for (int j = 0; j < 16; ++j) b2[j] = p2[j];
      #pragma unroll
      for (int j = 0; j < 16; ++j) b3[j] = p3[j];
      acc[0] = __builtin_amdgcn_wmma_f32_16x16x32_f16(false, a, false, b0, (short)0, acc[0], false, false);
      acc[1] = __builtin_amdgcn_wmma_f32_16x16x32_f16(false, a, false, b1, (short)0, acc[1], false, false);
      acc[2] = __builtin_amdgcn_wmma_f32_16x16x32_f16(false, a, false, b2, (short)0, acc[2], false, false);
      acc[3] = __builtin_amdgcn_wmma_f32_16x16x32_f16(false, a, false, b3, (short)0, acc[3], false, false);
    }

    __syncthreads();   // all waves done reading h_pad for this step
    if (lane < 16) {
      const int unit = u0 + nl;
      const float* xwt = xw + (size_t)(t * B_) * G4_;
      #pragma unroll
      for (int r = 0; r < 8; ++r) {        // r == batch index (rows 8..15 are pad)
        const float* xwr = xwt + (size_t)r * G4_;
        float iv = acc[0][r] + xwr[unit];
        float fv = acc[1][r] + xwr[H_ + unit];
        float gv = acc[2][r] + xwr[2 * H_ + unit];
        float ov = acc[3][r] + xwr[3 * H_ + unit];
        float outv = 0.0f;
        if (t < nlen[r]) {
          float cn = sigf(fv) * c_reg[r] + sigf(iv) * tanhf(gv);
          c_reg[r] = cn;
          h_reg[r] = sigf(ov) * tanhf(cn);
          outv = h_reg[r];
        }
        size_t o = (size_t)(t * B_ + r) * DH_ + (size_t)dir * H_ + unit;
        hs[o] = outv;
        hs_h[o] = (_Float16)outv;
        h_pad[(size_t)r * H_ + unit] = (_Float16)h_reg[r];
      }
    }
    __syncthreads();   // h_pad update visible before next step's A loads
  }
}

// ----------------------------------------------------------------------------
// LayerNorm over last dim (1024), then multiply by time mask
// ----------------------------------------------------------------------------
__global__ void layernorm_kernel(const float* __restrict__ hsrc,
                                 const float* __restrict__ gamma, const float* __restrict__ beta,
                                 const int* __restrict__ nl,
                                 float* __restrict__ hln, _Float16* __restrict__ hln_h) {
  const int r = blockIdx.x;            // t*B + b
  const int t = r / B_, b = r % B_;
  const int tid = threadIdx.x;
  __shared__ float red[256];
  const float* row = hsrc + (size_t)r * DH_;
  float s = 0.0f;
  for (int i = tid; i < DH_; i += 256) s += row[i];
  red[tid] = s; __syncthreads();
  for (int k = 128; k > 0; k >>= 1) { if (tid < k) red[tid] += red[tid + k]; __syncthreads(); }
  const float mean = red[0] / (float)DH_;
  __syncthreads();
  float v = 0.0f;
  for (int i = tid; i < DH_; i += 256) { float d = row[i] - mean; v += d * d; }
  red[tid] = v; __syncthreads();
  for (int k = 128; k > 0; k >>= 1) { if (tid < k) red[tid] += red[tid + k]; __syncthreads(); }
  const float inv = rsqrtf(red[0] / (float)DH_ + 1e-5f);
  const float m = (t < nl[b]) ? 1.0f : 0.0f;
  for (int i = tid; i < DH_; i += 256) {
    float o = ((row[i] - mean) * inv * gamma[i] + beta[i]) * m;
    hln[(size_t)r * DH_ + i] = o;
    hln_h[(size_t)r * DH_ + i] = (_Float16)o;
  }
}

// ----------------------------------------------------------------------------
// Decoder: attention energy  e[b][t]
// ----------------------------------------------------------------------------
__global__ void dec_energy_kernel(const float* __restrict__ alpha,
                                  const float* __restrict__ conv_w,
                                  const float* __restrict__ W_fe,
                                  const float* __restrict__ W_ee,
                                  const float* __restrict__ sW,
                                  const float* __restrict__ hW,
                                  float* __restrict__ e_out) {
  const int b = blockIdx.y;
  const int t = blockIdx.x * blockDim.x + threadIdx.x;
  if (t >= T2_) return;
  float convc[10];
  #pragma unroll
  for (int c = 0; c < 10; ++c) {
    float s = 0.0f;
    for (int k = 0; k < 100; ++k) {
      int tt = t - 50 + k;
      if (tt >= 0 && tt < T2_) s += alpha[b * T2_ + tt] * conv_w[c * 100 + k];
    }
    convc[c] = s;
  }
  float acc = 0.0f;
  const float* sWb = sW + (size_t)b * DH_;
  const float* hWr = hW + (size_t)(t * B_ + b) * DH_;
  for (int e = 0; e < DH_; ++e) {
    float cf = 0.0f;
    #pragma unroll
    for (int c = 0; c < 10; ++c) cf += convc[c] * W_fe[e * 10 + c];
    acc += tanhf(sWb[e] + hWr[e] + cf) * W_ee[e];
  }
  e_out[b * T2_ + t] = acc;
}

// ----------------------------------------------------------------------------
// Decoder: masked softmax over t + context g = sum_t a*h_ln ; writes alpha
// ----------------------------------------------------------------------------
__global__ void dec_softmax_ctx_kernel(const float* __restrict__ e, const float* __restrict__ hln,
                                       const int* __restrict__ nl,
                                       float* __restrict__ alpha,
                                       float* __restrict__ g, _Float16* __restrict__ g_h) {
  const int b = blockIdx.x, tid = threadIdx.x;
  __shared__ float red[256];
  float mx = -3.4e38f;
  for (int t = tid; t < T2_; t += 256) mx = fmaxf(mx, e[b * T2_ + t]);
  red[tid] = mx; __syncthreads();
  for (int k = 128; k > 0; k >>= 1) { if (tid < k) red[tid] = fmaxf(red[tid], red[tid + k]); __syncthreads(); }
  mx = red[0]; __syncthreads();
  const int nlb = nl[b];
  float sum = 0.0f;
  for (int t = tid; t < T2_; t += 256) {
    float en = (t < nlb) ? expf(e[b * T2_ + t] - mx) : 0.0f;
    alpha[b * T2_ + t] = en;
    sum += en;
  }
  red[tid] = sum; __syncthreads();
  for (int k = 128; k > 0; k >>= 1) { if (tid < k) red[tid] += red[tid + k]; __syncthreads(); }
  const float inv = 1.0f / red[0];
  __syncthreads();
  for (int t = tid; t < T2_; t += 256) alpha[b * T2_ + t] *= inv;
  __syncthreads();
  #pragma unroll
  for (int j = 0; j < 4; ++j) {
    const int d = tid + j * 256;
    float acc = 0.0f;
    for (int t = 0; t < T2_; ++t) acc += alpha[b * T2_ + t] * hln[(size_t)(t * B_ + b) * DH_ + d];
    g[(size_t)b * DH_ + d] = acc;
    g_h[(size_t)b * DH_ + d] = (_Float16)acc;
  }
}

// ----------------------------------------------------------------------------
// Decoder: t = tanh(g@W_gy^T + s2@W_sy^T + b_gy)
// ----------------------------------------------------------------------------
__global__ void dec_y1_kernel(const float* __restrict__ g, const float* __restrict__ s2,
                              const float* __restrict__ W_gy, const float* __restrict__ W_sy,
                              const float* __restrict__ b_gy,
                              float* __restrict__ tvec, _Float16* __restrict__ tvec_h) {
  int idx = blockIdx.x * blockDim.x + threadIdx.x;
  if (idx >= B_ * H_) return;
  int b = idx / H_, n = idx % H_;
  float acc = b_gy[n];
  const float* gb = g + (size_t)b * DH_;
  const float* wg = W_gy + (size_t)n * DH_;
  for (int k = 0; k < DH_; ++k) acc += gb[k] * wg[k];
  const float* sb = s2 + (size_t)b * H_;
  const float* ws = W_sy + (size_t)n * H_;
  for (int k = 0; k < H_; ++k) acc += sb[k] * ws[k];
  float v = tanhf(acc);
  tvec[idx] = v; tvec_h[idx] = (_Float16)v;
}

// ----------------------------------------------------------------------------
// Decoder: rec = [emb] + x1@W1^T + x2@W2^T + x3@W3^T + bias  (N = 2048)
// ----------------------------------------------------------------------------
__global__ void dec_rec_kernel(const float* x1, const float* W1, int k1,
                               const float* x2, const float* W2, int k2,
                               const float* x3, const float* W3, int k3,
                               const float* __restrict__ bias, const float* emb,
                               const int* target, int step, float* __restrict__ rec) {
  int idx = blockIdx.x * blockDim.x + threadIdx.x;
  if (idx >= B_ * G4_) return;
  int b = idx / G4_, n = idx % G4_;
  float acc = bias[n];
  if (emb) acc += emb[(size_t)target[b * U_ + step] * G4_ + n];
  if (x1) { const float* xb = x1 + (size_t)b * k1; const float* w = W1 + (size_t)n * k1;
            for (int k = 0; k < k1; ++k) acc += xb[k] * w[k]; }
  if (x2) { const float* xb = x2 + (size_t)b * k2; const float* w = W2 + (size_t)n * k2;
            for (int k = 0; k < k2; ++k) acc += xb[k] * w[k]; }
  if (x3) { const float* xb = x3 + (size_t)b * k3; const float* w = W3 + (size_t)n * k3;
            for (int k = 0; k < k3; ++k) acc += xb[k] * w[k]; }
  rec[idx] = acc;
}

__global__ void dec_cell_kernel(const float* __restrict__ rec,
                                float* __restrict__ s, float* __restrict__ c,
                                _Float16* s_h) {
  int idx = blockIdx.x * blockDim.x + threadIdx.x;
  if (idx >= B_ * H_) return;
  int b = idx / H_, u = idx % H_;
  const float* rb = rec + (size_t)b * G4_;
  float iv = rb[u], fv = rb[H_ + u], gv = rb[2 * H_ + u], ov = rb[3 * H_ + u];
  float cn = sigf(fv) * c[idx] + sigf(iv) * tanhf(gv);
  c[idx] = cn;
  float hn = sigf(ov) * tanhf(cn);
  s[idx] = hn;
  if (s_h) s_h[idx] = (_Float16)hn;
}

// ----------------------------------------------------------------------------
// host orchestration
// ----------------------------------------------------------------------------
extern "C" void kernel_launch(void* const* d_in, const int* in_sizes, int n_in,
                              void* d_out, int out_size, void* d_ws, size_t ws_size,
                              hipStream_t stream) {
  (void)in_sizes; (void)n_in; (void)out_size; (void)ws_size;
  const float* speech   = (const float*)d_in[0];
  const int*   lengths  = (const int*)d_in[1];
  const int*   target   = (const int*)d_in[2];
  const float* conv1_w  = (const float*)d_in[3];
  const float* conv1_b  = (const float*)d_in[4];
  const float* bn1_g    = (const float*)d_in[5];
  const float* bn1_b    = (const float*)d_in[6];
  const float* bn1_m    = (const float*)d_in[7];
  const float* bn1_v    = (const float*)d_in[8];
  const float* conv2_w  = (const float*)d_in[9];
  const float* conv2_b  = (const float*)d_in[10];
  const float* bn2_g    = (const float*)d_in[11];
  const float* bn2_b    = (const float*)d_in[12];
  const float* bn2_m    = (const float*)d_in[13];
  const float* bn2_v    = (const float*)d_in[14];
  const float* Wih0     = (const float*)d_in[15];
  const float* Whh0     = (const float*)d_in[16];
  const float* b0       = (const float*)d_in[17];
  const float* WihL     = (const float*)d_in[18];
  const float* WhhL     = (const float*)d_in[19];
  const float* bL       = (const float*)d_in[20];
  const float* ln_gamma = (const float*)d_in[21];
  const float* ln_beta  = (const float*)d_in[22];
  const float* W_se     = (const float*)d_in[23];
  const float* W_he     = (const float*)d_in[24];
  const float* b_he     = (const float*)d_in[25];
  const float* W_ee     = (const float*)d_in[26];
  const float* conv_att = (const float*)d_in[27];
  const float* W_fe     = (const float*)d_in[28];
  const float* W_sy     = (const float*)d_in[29];
  const float* W_gy     = (const float*)d_in[30];
  const float* b_gy     = (const float*)d_in[31];
  const float* W_yy     = (const float*)d_in[32];
  const float* b_yy     = (const float*)d_in[33];
  const float* emb_ys   = (const float*)d_in[34];
  const float* W_ss1    = (const float*)d_in[35];
  const float* W_gs1    = (const float*)d_in[36];
  const float* b_gs1    = (const float*)d_in[37];
  const float* W_ss12   = (const float*)d_in[38];
  const float* W_ss2    = (const float*)d_in[39];
  const float* W_gs2    = (const float*)d_in[40];
  const float* b_gs2    = (const float*)d_in[41];
  float* out = (float*)d_out;

  char* base = (char*)d_ws;
  size_t off = 0;
  auto alloc = [&](size_t bytes) -> void* {
    off = (off + 255) & ~(size_t)255;
    void* p = base + off;
    off += bytes;
    return p;
  };

  const size_t MROWS = (size_t)T2_ * B_;           // 2392
  // f16 weights
  _Float16* wih0_h = (_Float16*)alloc((size_t)2 * G4_ * 320 * 2);
  _Float16* whh_h  = (_Float16*)alloc((size_t)8 * G4_ * H_ * 2);
  _Float16* wihL_h = (_Float16*)alloc((size_t)6 * G4_ * DH_ * 2);
  _Float16* wse_h  = (_Float16*)alloc((size_t)DH_ * H_ * 2);
  _Float16* whe_h  = (_Float16*)alloc((size_t)DH_ * DH_ * 2);
  _Float16* wyy_h  = (_Float16*)alloc((size_t)C_ * H_ * 2);
  // activations
  float*    a1     = (float*)alloc((size_t)B_ * 32 * 20 * 599 * 4);
  float*    cnn    = (float*)alloc(MROWS * 320 * 4);
  _Float16* cnn_h  = (_Float16*)alloc(MROWS * 320 * 2);
  float*    xw_fw  = (float*)alloc(MROWS * G4_ * 4);
  float*    xw_bw  = (float*)alloc(MROWS * G4_ * 4);
  float*    hsbuf  = (float*)alloc(MROWS * DH_ * 4);
  _Float16* hs_h   = (_Float16*)alloc(MROWS * DH_ * 2);
  float*    h_ln   = (float*)alloc(MROWS * DH_ * 4);
  _Float16* hln_h  = (_Float16*)alloc(MROWS * DH_ * 2);
  float*    hW     = (float*)alloc(MROWS * DH_ * 4);
  // decoder
  float*    sWb    = (float*)alloc((size_t)B_ * DH_ * 4);
  float*    e_buf  = (float*)alloc((size_t)B_ * T2_ * 4);
  float*    alpha  = (float*)alloc((size_t)B_ * T2_ * 4);
  float*    gctx   = (float*)alloc((size_t)B_ * DH_ * 4);
  _Float16* g_h    = (_Float16*)alloc((size_t)B_ * DH_ * 2);
  float*    tvec   = (float*)alloc((size_t)B_ * H_ * 4);
  _Float16* tvec_h = (_Float16*)alloc((size_t)B_ * H_ * 2);
  float*    rec    = (float*)alloc((size_t)B_ * G4_ * 4);
  float*    s1     = (float*)alloc((size_t)B_ * H_ * 4);
  float*    c1     = (float*)alloc((size_t)B_ * H_ * 4);
  float*    s2     = (float*)alloc((size_t)B_ * H_ * 4);
  float*    c2     = (float*)alloc((size_t)B_ * H_ * 4);
  _Float16* s1_h   = (_Float16*)alloc((size_t)B_ * H_ * 2);
  int*      newlen = (int*)alloc(B_ * 4);

  auto cvt = [&](const float* s, _Float16* d, size_t n) {
    cvt_f16_kernel<<<dim3((unsigned)((n + 255) / 256)), 256, 0, stream>>>(s, d, (int)n);
  };
  auto gemm = [&](const _Float16* A, int lda, const _Float16* W, int ldw,
                  float* C, int ldc, _Float16* Ch, int ldch, const float* bias,
                  int M, int N, int K) {
    dim3 grid((N + 511) / 512, (M + 15) / 16);
    gemm_wmma_kernel<<<grid, 256, 0, stream>>>(A, lda, W, ldw, C, ldc, Ch, ldch, bias, M, N, K);
  };

  // ---- weight conversion (f32 -> f16, L2-resident thereafter) ----
  cvt(Wih0, wih0_h, (size_t)2 * G4_ * 320);
  cvt(Whh0, whh_h, (size_t)2 * G4_ * H_);
  cvt(WhhL, whh_h + (size_t)2 * G4_ * H_, (size_t)6 * G4_ * H_);
  cvt(WihL, wihL_h, (size_t)6 * G4_ * DH_);
  cvt(W_se, wse_h, (size_t)DH_ * H_);
  cvt(W_he, whe_h, (size_t)DH_ * DH_);
  cvt(W_yy, wyy_h, (size_t)C_ * H_);

  newlen_kernel<<<1, 32, 0, stream>>>(lengths, newlen);

  // ---- conv front-end ----
  {
    int n1 = B_ * 32 * 20 * 599;
    conv1_kernel<<<(n1 + 255) / 256, 256, 0, stream>>>(speech, conv1_w, conv1_b,
                                                       bn1_g, bn1_b, bn1_m, bn1_v, a1);
    int n2 = B_ * 32 * 10 * T2_;
    conv2_kernel<<<(n2 + 255) / 256, 256, 0, stream>>>(a1, conv2_w, conv2_b,
                                                       bn2_g, bn2_b, bn2_m, bn2_v, cnn, cnn_h);
  }

  // ---- 4 BiLSTM layers ----
  const _Float16* x_h = cnn_h;
  int Kin = 320;
  for (int l = 0; l < 4; ++l) {
    const _Float16* whf = whh_h + (size_t)(l * 2 + 0) * G4_ * H_;
    const _Float16* whb = whh_h + (size_t)(l * 2 + 1) * G4_ * H_;
    const _Float16 *wf, *wb;
    const float *bf, *bb;
    if (l == 0) {
      wf = wih0_h;                     wb = wih0_h + (size_t)G4_ * 320;
      bf = b0;                         bb = b0 + G4_;
    } else {
      wf = wihL_h + (size_t)((l - 1) * 2) * G4_ * DH_;
      wb = wf + (size_t)G4_ * DH_;
      bf = bL + (size_t)((l - 1) * 2) * G4_;
      bb = bf + G4_;
    }
    gemm(x_h, Kin, wf, Kin, xw_fw, G4_, nullptr, 0, bf, (int)MROWS, G4_, Kin);
    gemm(x_h, Kin, wb, Kin, xw_bw, G4_, nullptr, 0, bb, (int)MROWS, G4_, Kin);
    lstm_scan_kernel<<<2, 1024, 0, stream>>>(xw_fw, xw_bw, whf, whb, newlen, hsbuf, hs_h);
    x_h = hs_h;
    Kin = DH_;
  }

  // ---- LayerNorm + mask, and hW = h_ln @ W_he^T + b_he ----
  layernorm_kernel<<<(unsigned)MROWS, 256, 0, stream>>>(hsbuf, ln_gamma, ln_beta, newlen, h_ln, hln_h);
  gemm(hln_h, DH_, whe_h, DH_, hW, DH_, nullptr, 0, b_he, (int)MROWS, DH_, DH_);

  // ---- decoder init ----
  zero_f32_kernel<<<(B_ * H_ + 255) / 256, 256, 0, stream>>>(s1, B_ * H_);
  zero_f32_kernel<<<(B_ * H_ + 255) / 256, 256, 0, stream>>>(c1, B_ * H_);
  zero_f32_kernel<<<(B_ * H_ + 255) / 256, 256, 0, stream>>>(s2, B_ * H_);
  zero_f32_kernel<<<(B_ * H_ + 255) / 256, 256, 0, stream>>>(c2, B_ * H_);
  zero_f32_kernel<<<(B_ * T2_ + 255) / 256, 256, 0, stream>>>(alpha, B_ * T2_);
  zero_f16_kernel<<<(B_ * H_ + 255) / 256, 256, 0, stream>>>(s1_h, B_ * H_);

  // ---- 40 decoder steps ----
  for (int step = 0; step < U_; ++step) {
    gemm(s1_h, H_, wse_h, H_, sWb, DH_, nullptr, 0, nullptr, B_, DH_, H_);
    dec_energy_kernel<<<dim3(3, B_), 128, 0, stream>>>(alpha, conv_att, W_fe, W_ee, sWb, hW, e_buf);
    dec_softmax_ctx_kernel<<<B_, 256, 0, stream>>>(e_buf, h_ln, newlen, alpha, gctx, g_h);
    dec_y1_kernel<<<(B_ * H_ + 255) / 256, 256, 0, stream>>>(gctx, s2, W_gy, W_sy, b_gy, tvec, tvec_h);
    // y = tanh(..) @ W_yy^T + b_yy  -> out[b][step][:]
    gemm(tvec_h, H_, wyy_h, H_, out + (size_t)step * C_, U_ * C_, nullptr, 0, b_yy, B_, C_, H_);
    // LSTM cell 1
    dec_rec_kernel<<<(B_ * G4_ + 255) / 256, 256, 0, stream>>>(
        s1, W_ss1, H_, gctx, W_gs1, DH_, nullptr, nullptr, 0, b_gs1, emb_ys, target, step, rec);
    dec_cell_kernel<<<(B_ * H_ + 255) / 256, 256, 0, stream>>>(rec, s1, c1, s1_h);
    // LSTM cell 2
    dec_rec_kernel<<<(B_ * G4_ + 255) / 256, 256, 0, stream>>>(
        s1, W_ss12, H_, s2, W_ss2, H_, gctx, W_gs2, DH_, b_gs2, nullptr, nullptr, 0, rec);
    dec_cell_kernel<<<(B_ * H_ + 255) / 256, 256, 0, stream>>>(rec, s2, c2, nullptr);
  }
}
